// AugmentedAffinityContrastive_56882546868661
// MI455X (gfx1250) — matrix operand order repaired and checked by reading, not compiled, not run on hardware
//
#include <hip/hip_runtime.h>

typedef __attribute__((ext_vector_type(16))) _Float16 v16h;
typedef __attribute__((ext_vector_type(8)))  float    v8f;

#define HWSZ 262144u           // 512*512
#define NWAVE 65536            // 4 * 512 * 32

// ---------------- init: per-offset min/max cells ----------------
__global__ void k_init(unsigned* __restrict__ mm) {
  int t = threadIdx.x;
  if (t < 9) { mm[2 * t] = 0x7F800000u; mm[2 * t + 1] = 0u; }
}

// ---------------- Gaussian blur, sigma=1.2, radius=5, edge clamp ----------------
__device__ __forceinline__ void gauss_w(float* kw) {
  float s = 0.f;
#pragma unroll
  for (int t = 0; t < 11; ++t) {
    float x = (float)(t - 5) * (1.0f / 1.2f);
    kw[t] = expf(-0.5f * x * x);
    s += kw[t];
  }
  float inv = 1.0f / s;
#pragma unroll
  for (int t = 0; t < 11; ++t) kw[t] *= inv;
}

__global__ void k_blurx(const float* __restrict__ src, float* __restrict__ dst) {
  int p = blockIdx.x * blockDim.x + threadIdx.x;
  int h = p >> 9, w = p & 511;
  float kw[11]; gauss_w(kw);
  float acc = 0.f;
#pragma unroll
  for (int t = 0; t < 11; ++t) {
    int ww = w + t - 5; ww = ww < 0 ? 0 : (ww > 511 ? 511 : ww);
    acc += kw[t] * src[(h << 9) + ww];
  }
  dst[p] = acc;
}

__global__ void k_blury(const float* __restrict__ src, float* __restrict__ dst) {
  int p = blockIdx.x * blockDim.x + threadIdx.x;
  int h = p >> 9, w = p & 511;
  float kw[11]; gauss_w(kw);
  float acc = 0.f;
#pragma unroll
  for (int t = 0; t < 11; ++t) {
    int hh = h + t - 5; hh = hh < 0 ? 0 : (hh > 511 ? 511 : hh);
    acc += kw[t] * src[(hh << 9) + w];
  }
  dst[p] = acc;
}

// ---------------- per-offset |d| min/max (uint order == float order, d>=0) ----------------
__global__ void k_minmax(const float* __restrict__ bl, unsigned* __restrict__ mm) {
  const int DY[9] = {0, 1, 1, 0, 2, 2, 0, 3, 3};
  const int DX[9] = {1, 0, 1, 2, 0, 2, 3, 0, 3};
  int i = blockIdx.y;
  int p = blockIdx.x * blockDim.x + threadIdx.x;
  int h = p >> 9, w = p & 511;
  float d = fabsf(bl[p] - bl[(((h - DY[i]) & 511) << 9) + ((w - DX[i]) & 511)]);
  unsigned du = __float_as_uint(d);
  __shared__ unsigned smn, smx;
  if (threadIdx.x == 0) { smn = 0x7F800000u; smx = 0u; }
  __syncthreads();
  atomicMin(&smn, du);
  atomicMax(&smx, du);
  __syncthreads();
  if (threadIdx.x == 0) {
    atomicMin(&mm[2 * i], smn);
    atomicMax(&mm[2 * i + 1], smx);
  }
}

// ---------------- bake c_i = mask * (0.5 - (d-min)/(max-min)) / (B*H*W) ----------------
__global__ void k_cplane(const float* __restrict__ bl, const float* __restrict__ mask,
                         const unsigned* __restrict__ mm, float* __restrict__ cpl) {
  const int DY[9] = {0, 1, 1, 0, 2, 2, 0, 3, 3};
  const int DX[9] = {1, 0, 1, 2, 0, 2, 3, 0, 3};
  int i = blockIdx.y;
  int p = blockIdx.x * blockDim.x + threadIdx.x;
  int h = p >> 9, w = p & 511;
  float d = fabsf(bl[p] - bl[(((h - DY[i]) & 511) << 9) + ((w - DX[i]) & 511)]);
  float mn = __uint_as_float(mm[2 * i]);
  float mx = __uint_as_float(mm[2 * i + 1]);
  float a = (d - mn) / (mx - mn);
  cpl[i * HWSZ + p] = mask[i * HWSZ + p] * (0.5f - a) * (1.0f / 1048576.0f);
}

// ---------------- csum = sum_i c_i  (absorbs the "+c" half of (1-D)*c) ----------------
__global__ void k_csum(const float* __restrict__ cpl, float* __restrict__ cs) {
  unsigned p = blockIdx.x * blockDim.x + threadIdx.x;
  float s = 0.f;
#pragma unroll
  for (unsigned i = 0; i < 9; ++i) s += cpl[i * HWSZ + p];
  cs[p] = s;
}

// ---------------- main loss: WMMA f32_16x16x32_f16, branchless band extraction ----------------
// nibble tables ((dy,dox)->offset-id map; 0xF = not present):
// dy=0: dx1->0 dx2->3 dx3->6 ; dy=1: dx0->1 dx1->2 ; dy=2: dx0->4 dx2->5 ; dy=3: dx0->7 dx3->8
__global__ void k_loss(const float* __restrict__ emb, const float* __restrict__ tfe,
                       const float* __restrict__ cpl, const float* __restrict__ cs,
                       float* __restrict__ partial) {
  int gt   = blockIdx.x * blockDim.x + threadIdx.x;
  int wave = gt >> 5;
  int lane = threadIdx.x & 31;
  unsigned b  = (unsigned)(wave >> 14);
  unsigned h  = (unsigned)((wave >> 5) & 511);
  unsigned w0 = (unsigned)((wave & 31) << 4);
  unsigned msel = (unsigned)(lane & 15);
  int      hi   = lane >> 4;           // lane group: selects K-half / M-half per ISA layout

  unsigned pix   = (h << 9) + w0;                  // 32-bit element indices everywhere:
  unsigned ebase = b * 32u * HWSZ + pix + msel;    // lets backend use SADDR+voffset loads
  unsigned cbase = pix + (hi ? 8u : 0u);           // + r -> pixel m = r + 8*hi

  // A[m,k] = embeds[b,k,h,w0+m]; A k order used here (same as B): lane<16 K 0..15, lane>=16 K 16..31
  v16h A;
#pragma unroll
  for (int j = 0; j < 16; ++j) {
    unsigned k = (unsigned)(j + (hi ? 16 : 0));
    A[j] = (_Float16)emb[ebase + k * HWSZ];
  }

  // seed accumulator with the "+c" term: lanes 0..15 pick up csum at the 16 tile pixels
  float cs0 = cs[pix + msel];
  float acc = hi ? 0.f : cs0;

  unsigned kb = hi ? 16u : 0u;         // B layout: lane<16 K 0..15, lane>=16 K 16..31
#pragma unroll
  for (int dy = 0; dy < 4; ++dy) {
    unsigned row = (h - (unsigned)dy) & 511u;
    unsigned tab = (dy < 2) ? 0xFF21630Fu : 0x8FF7F5F4u;   // compile-time per dy
#pragma unroll
    for (int blk = 0; blk < 2; ++blk) {
      unsigned col = (w0 - 3u + (unsigned)blk * 16u + msel) & 511u;
      unsigned tbase = (b * 32u + kb) * HWSZ + (row << 9) + col;
      v16h Bv;
#pragma unroll
      for (unsigned j = 0; j < 16; ++j)
        Bv[j] = (_Float16)tfe[tbase + j * HWSZ];

      v8f D = {0.f, 0.f, 0.f, 0.f, 0.f, 0.f, 0.f, 0.f};
      D = __builtin_amdgcn_wmma_f32_16x16x32_f16(false, A, false, Bv,
                                                 (short)0, D, false, false);
#pragma unroll
      for (int r = 0; r < 8; ++r) {
        if (blk == 1 && r < 5) continue;           // blk1 needs m>=13 -> r in {5,6,7}
        int m   = r + (hi ? 8 : 0);                // D: M = r + 8*(lane>=16), N = lane&15
        int dox = (blk == 0) ? (m + 3 - (int)msel) : (m - (int)msel - 13);
        unsigned sh  = ((unsigned)((((dy & 1) << 2) + dox) << 2)) & 31u;
        unsigned nib = (tab >> sh) & 15u;
        bool use = ((unsigned)dox <= 3u) && (nib != 15u);
        unsigned plane = use ? nib : 0u;           // always a valid plane
        float cv = cpl[cbase + plane * HWSZ + (unsigned)r];   // unconditional load
        cv = use ? cv : 0.0f;
        acc = fmaf(-D[r], cv, acc);                // accumulate -D*c
      }
    }
  }

#pragma unroll
  for (int s = 16; s > 0; s >>= 1) acc += __shfl_xor(acc, s, 32);
  if (lane == 0) partial[wave] = acc;
}

// ---------------- deterministic final reduction ----------------
__global__ void k_reduce(const float* __restrict__ partial, float* __restrict__ out) {
  __shared__ float sm[256];
  int t = threadIdx.x;
  float s = 0.f;
  for (int i = t; i < NWAVE; i += 256) s += partial[i];
  sm[t] = s;
  __syncthreads();
  for (int st = 128; st > 0; st >>= 1) {
    if (t < st) sm[t] += sm[t + st];
    __syncthreads();
  }
  if (t == 0) out[0] = sm[0];
}

extern "C" void kernel_launch(void* const* d_in, const int* in_sizes, int n_in,
                              void* d_out, int out_size, void* d_ws, size_t ws_size,
                              hipStream_t stream) {
  (void)in_sizes; (void)n_in; (void)out_size; (void)ws_size;
  const float* emb  = (const float*)d_in[0];   // [4,32,512,512]
  const float* tfe  = (const float*)d_in[1];   // [4,32,512,512]
  const float* raw  = (const float*)d_in[2];   // [1,1,512,512]
  const float* mask = (const float*)d_in[3];   // [1,9,512,512]
  float* out = (float*)d_out;

  float* wsf     = (float*)d_ws;
  float* bt      = wsf;                          // 512*512
  float* bl      = wsf + HWSZ;                   // 512*512
  float* cpl     = wsf + 2 * HWSZ;               // 9 * 512*512
  float* partial = wsf + 11 * HWSZ;              // 65536
  float* cs      = wsf + 11 * HWSZ + NWAVE;      // 512*512
  unsigned* mm   = (unsigned*)(wsf + 12 * HWSZ + NWAVE);  // 18 uints

  k_init  <<<1, 32, 0, stream>>>(mm);
  k_blurx <<<1024, 256, 0, stream>>>(raw, bt);
  k_blury <<<1024, 256, 0, stream>>>(bt, bl);
  k_minmax<<<dim3(1024, 9), 256, 0, stream>>>(bl, mm);
  k_cplane<<<dim3(1024, 9), 256, 0, stream>>>(bl, mask, mm, cpl);
  k_csum  <<<1024, 256, 0, stream>>>(cpl, cs);
  k_loss  <<<8192, 256, 0, stream>>>(emb, tfe, cpl, cs, partial);
  k_reduce<<<1, 256, 0, stream>>>(partial, out);
}
